// HAM_21741124452976
// MI455X (gfx1250) — matrix operand order repaired and checked by reading, not compiled, not run on hardware
//
#include <hip/hip_runtime.h>

typedef __attribute__((ext_vector_type(16))) _Float16 v16h;
typedef __attribute__((ext_vector_type(8)))  _Float16 v8h;
typedef __attribute__((ext_vector_type(8)))  float    v8f;

#define NB    4
#define LSEQ  2304        // 48*48
#define CS    128         // channels per scale == D_MODEL
#define DI    256         // D_INNER
#define NST   16          // D_STATE
#define MROWS (NB*LSEQ)   // 9216

__device__ __forceinline__ float fsig(float x)  { return 1.f / (1.f + __expf(-x)); }
__device__ __forceinline__ float fsilu(float x) { return x * fsig(x); }

__device__ __forceinline__ v8f wmma_f16(v16h a, v16h b, v8f c) {
    return __builtin_amdgcn_wmma_f32_16x16x32_f16(false, a, false, b, (short)0, c, false, false);
}

// ---- stage 0: x_i = spx[i] (+ prev), and pack f16 A-matrix u[(b*L+l)*128 + d]
__global__ void prep_kernel(const float* __restrict__ x, const float* __restrict__ prev,
                            float* __restrict__ xi, _Float16* __restrict__ uA,
                            int scale, int usePrev) {
    int idx = blockIdx.x * blockDim.x + threadIdx.x;       // (b, d, l), l fastest
    if (idx >= NB * CS * LSEQ) return;
    int l = idx % LSEQ;
    int d = (idx / LSEQ) % CS;
    int b = idx / (LSEQ * CS);
    float v = x[((size_t)b * 512 + scale * CS + d) * LSEQ + l];
    if (usePrev) v += prev[idx];
    xi[idx] = v;
    uA[((size_t)(b * LSEQ + l)) * CS + d] = (_Float16)v;
}

// ---- weight f32 -> f16, with optional zero row padding (Wx 40 -> 48)
__global__ void cvt_w_kernel(const float* __restrict__ src, _Float16* __restrict__ dst,
                             int srcRows, int dstRows, int cols) {
    int idx = blockIdx.x * blockDim.x + threadIdx.x;
    if (idx >= dstRows * cols) return;
    int r = idx / cols;
    dst[idx] = (r < srcRows) ? (_Float16)src[idx] : (_Float16)0.f;
}

// ---- register-blocked WMMA GEMM: C[M,N] f32 = A[M,K]f16 @ Bw[N,K]f16^T
// one wave per 32(M) x 64(N) strip: 2 A-frags x 4 B-frags -> 8 independent WMMAs per K step.
// Requires M%32==0, N%64==0, K%32==0.
__global__ __launch_bounds__(256)
void wmma_gemm_blk_kernel(const _Float16* __restrict__ A, const _Float16* __restrict__ Bw,
                          float* __restrict__ C, int Mt, int Nt, int K) {
    int wave = (blockIdx.x * blockDim.x + threadIdx.x) >> 5;   // uniform per wave
    int lane = threadIdx.x & 31;
    int wN = Nt >> 6;
    int numW = (Mt >> 5) * wN;
    if (wave >= numW) return;
    int tm = wave / wN, tn = wave % wN;
    int hh  = lane >> 4;
    int sub = lane & 15;
    const _Float16* arow0 = A + (size_t)(tm * 32 + sub) * K + hh * 16;
    const _Float16* arow1 = arow0 + (size_t)16 * K;
    const _Float16* bcol  = Bw + (size_t)(tn * 64 + sub) * K + hh * 16;
    v8f acc[2][4] = {};
    for (int k = 0; k < K; k += 32) {
        __builtin_prefetch(arow0 + k + 128, 0, 1);
        __builtin_prefetch(arow1 + k + 128, 0, 1);
        v8h t0 = *(const v8h*)(arow0 + k);
        v8h t1 = *(const v8h*)(arow0 + k + 8);
        v8h t2 = *(const v8h*)(arow1 + k);
        v8h t3 = *(const v8h*)(arow1 + k + 8);
        v16h a0, a1, bv[4];
#pragma unroll
        for (int j = 0; j < 8; ++j) {
            a0[j] = t0[j]; a0[8 + j] = t1[j];
            a1[j] = t2[j]; a1[8 + j] = t3[j];
        }
#pragma unroll
        for (int q = 0; q < 4; ++q)
            bv[q] = *(const v16h*)(bcol + (size_t)q * 16 * K + k);
#pragma unroll
        for (int q = 0; q < 4; ++q) {
            acc[0][q] = wmma_f16(a0, bv[q], acc[0][q]);
            acc[1][q] = wmma_f16(a1, bv[q], acc[1][q]);
        }
    }
#pragma unroll
    for (int i = 0; i < 2; ++i) {
        int rowb = tm * 32 + i * 16 + hh * 8;   // VGPR r: lanes0-15 -> M=r, lanes16-31 -> M=r+8
#pragma unroll
        for (int q = 0; q < 4; ++q) {
            int col = tn * 64 + q * 16 + sub;
#pragma unroll
            for (int r = 0; r < 8; ++r)
                C[(size_t)(rowb + r) * Nt + col] = acc[i][q][r];
        }
    }
}

// ---- single-tile WMMA GEMM for small N (dbc: N=48)
__global__ __launch_bounds__(256)
void wmma_gemm_kernel(const _Float16* __restrict__ A, const _Float16* __restrict__ Bw,
                      float* __restrict__ C, int Mt, int Nt, int K) {
    int wave = (blockIdx.x * blockDim.x + threadIdx.x) >> 5;
    int lane = threadIdx.x & 31;
    int tilesN = Nt >> 4;
    int numTiles = (Mt >> 4) * tilesN;
    if (wave >= numTiles) return;
    int tm = wave / tilesN, tn = wave % tilesN;
    int hh  = lane >> 4;
    int sub = lane & 15;
    const _Float16* arow = A  + (size_t)(tm * 16 + sub) * K + hh * 16;
    const _Float16* bcol = Bw + (size_t)(tn * 16 + sub) * K + hh * 16;
    v8f acc = {};
    for (int k = 0; k < K; k += 32) {
        __builtin_prefetch(arow + k + 128, 0, 1);
        v8h a0 = *(const v8h*)(arow + k);
        v8h a1 = *(const v8h*)(arow + k + 8);
        v16h av, bv;
        bv = *(const v16h*)(bcol + k);
#pragma unroll
        for (int j = 0; j < 8; ++j) { av[j] = a0[j]; av[8 + j] = a1[j]; }
        acc = wmma_f16(av, bv, acc);
    }
    int col  = tn * 16 + sub;
    int rowb = tm * 16 + hh * 8;
#pragma unroll
    for (int r = 0; r < 8; ++r) C[(size_t)(rowb + r) * Nt + col] = acc[r];
}

// ---- depthwise causal conv (k=4) + bias + SiLU over xp = xz[:, :256]
__global__ void conv_silu_kernel(const float* __restrict__ xz, const float* __restrict__ wc,
                                 const float* __restrict__ bc, float* __restrict__ xs,
                                 _Float16* __restrict__ xsH) {
    int idx = blockIdx.x * blockDim.x + threadIdx.x;   // (m, d), d fastest
    if (idx >= MROWS * DI) return;
    int d = idx % DI;
    int m = idx / DI;
    int l = m % LSEQ, b = m / LSEQ;
    float acc = bc[d];
#pragma unroll
    for (int j = 0; j < 4; ++j) {
        int lt = l - 3 + j;
        if (lt >= 0) acc += xz[((size_t)(b * LSEQ + lt)) * 512 + d] * wc[d * 4 + j];
    }
    float s = fsilu(acc);
    xs[idx]  = s;
    xsH[idx] = (_Float16)s;
}

// ---- delta = softplus(dt @ Wdt^T + bdt), K=8 (too small for WMMA)
__global__ void delta_kernel(const float* __restrict__ dbc, const float* __restrict__ Wdt,
                             const float* __restrict__ bdt, float* __restrict__ delta) {
    int idx = blockIdx.x * blockDim.x + threadIdx.x;   // (m, d), d fastest
    if (idx >= MROWS * DI) return;
    int d = idx % DI;
    int m = idx / DI;
    float acc = bdt[d];
#pragma unroll
    for (int r = 0; r < 8; ++r) acc += dbc[(size_t)m * 48 + r] * Wdt[d * 8 + r];
    delta[idx] = (acc > 20.f) ? acc : log1pf(__expf(acc));
}

// ---- selective scan: one lane per (b, d, n); 16-lane shuffle reduction over n.
__global__ __launch_bounds__(256)
void scan_kernel(const float* __restrict__ delta, const float* __restrict__ xs,
                 const float* __restrict__ dbc, const float* __restrict__ xz,
                 const float* __restrict__ A_log, const float* __restrict__ Dp,
                 _Float16* __restrict__ yh) {
    int gid  = blockIdx.x * blockDim.x + threadIdx.x;
    int lane = threadIdx.x & 31;
    int wave = gid >> 5;                 // 0..511
    int b = wave >> 7;
    int dpair = wave & 127;
    int d = dpair * 2 + (lane >> 4);
    int n = lane & 15;
    float Ad = -__expf(A_log[d * NST + n]);
    float Dd = Dp[d];
    float h = 0.f;
    for (int t = 0; t < LSEQ; ++t) {
        size_t m = (size_t)b * LSEQ + t;
        float dv = delta[m * DI + d];
        float xv = xs[m * DI + d];
        float Bv = dbc[m * 48 + 8  + n];
        float Cv = dbc[m * 48 + 24 + n];
        float dA = __expf(dv * Ad);                 // v_exp_f32, the dependent chain
        h = dA * h + dv * xv * Bv;
        float p = h * Cv;
        p += __shfl_xor(p, 8, 32);                  // stays inside the 16-lane group
        p += __shfl_xor(p, 4, 32);
        p += __shfl_xor(p, 2, 32);
        p += __shfl_xor(p, 1, 32);
        if (n == 0) {
            float z = xz[m * 512 + 256 + d];
            float y = (p + xv * Dd) * fsilu(z);
            yh[m * DI + d] = (_Float16)y;
        }
    }
}

// ---- cdgf: per-(b,c) means -> sigmoid gates
__global__ void cdgf_reduce_kernel(const float* __restrict__ xi, const float* __restrict__ xm,
                                   float* __restrict__ wred) {
    int bc = blockIdx.x;                // NB*CS blocks
    int c = bc % CS, b = bc / CS;
    float sl = 0.f, sg = 0.f;
    for (int l = threadIdx.x; l < LSEQ; l += blockDim.x) {
        float vi = xi[((size_t)b * CS + c) * LSEQ + l];
        float vm = xm[((size_t)(b * LSEQ + l)) * CS + c];
        sl += vi;
        sg += fmaxf(vi, vm);
    }
    __shared__ float s0[256], s1[256];
    s0[threadIdx.x] = sl; s1[threadIdx.x] = sg;
    __syncthreads();
    for (int st = 128; st > 0; st >>= 1) {
        if ((int)threadIdx.x < st) {
            s0[threadIdx.x] += s0[threadIdx.x + st];
            s1[threadIdx.x] += s1[threadIdx.x + st];
        }
        __syncthreads();
    }
    if (threadIdx.x == 0) {
        wred[bc * 2 + 0] = fsig(s0[0] / (float)LSEQ);
        wred[bc * 2 + 1] = fsig(s1[0] / (float)LSEQ);
    }
}

__global__ void cdgf_apply_kernel(const float* __restrict__ xi, const float* __restrict__ xm,
                                  const float* __restrict__ wred, float* __restrict__ out,
                                  float* __restrict__ prev, int scale) {
    int idx = blockIdx.x * blockDim.x + threadIdx.x;   // (b, c, l), l fastest
    if (idx >= NB * CS * LSEQ) return;
    int l = idx % LSEQ;
    int c = (idx / LSEQ) % CS;
    int b = idx / (LSEQ * CS);
    float vi = xi[idx];
    float vm = xm[((size_t)(b * LSEQ + l)) * CS + c];
    float fg = fmaxf(vi, vm);
    float f = wred[(b * CS + c) * 2 + 0] * vi + wred[(b * CS + c) * 2 + 1] * fg;
    out[((size_t)b * 512 + scale * CS + c) * LSEQ + l] = f;
    prev[idx] = f;
}

extern "C" void kernel_launch(void* const* d_in, const int* in_sizes, int n_in,
                              void* d_out, int out_size, void* d_ws, size_t ws_size,
                              hipStream_t stream) {
    (void)in_sizes; (void)n_in; (void)out_size; (void)ws_size;
    const float* x    = (const float*)d_in[0];
    const float* Wi   = (const float*)d_in[1];
    const float* wc   = (const float*)d_in[2];
    const float* bc   = (const float*)d_in[3];
    const float* Wx   = (const float*)d_in[4];
    const float* Wdt  = (const float*)d_in[5];
    const float* bdt  = (const float*)d_in[6];
    const float* Alog = (const float*)d_in[7];
    const float* Dp   = (const float*)d_in[8];
    const float* Wo   = (const float*)d_in[9];
    float* out = (float*)d_out;

    char* ws = (char*)d_ws;
    size_t off = 0;
    auto take = [&](size_t bytes) -> char* {
        char* p = ws + off;
        off += (bytes + 255) & ~(size_t)255;
        return p;
    };
    float*    xi    = (float*)take(sizeof(float) * NB * CS * LSEQ);
    float*    prev  = (float*)take(sizeof(float) * NB * CS * LSEQ);
    _Float16* uA    = (_Float16*)take(2ull * MROWS * CS);
    _Float16* wiH   = (_Float16*)take(2ull * 512 * 128);
    _Float16* wxH   = (_Float16*)take(2ull * 48 * 256);
    _Float16* woH   = (_Float16*)take(2ull * 128 * 256);
    float*    xz    = (float*)take(4ull * MROWS * 512);
    float*    xs    = (float*)take(4ull * MROWS * DI);
    _Float16* xsH   = (_Float16*)take(2ull * MROWS * DI);
    float*    dbc   = (float*)take(4ull * MROWS * 48);
    float*    delta = (float*)take(4ull * MROWS * DI);
    _Float16* yh    = (_Float16*)take(2ull * MROWS * DI);
    float*    xm    = (float*)take(4ull * MROWS * CS);
    float*    wred  = (float*)take(4ull * NB * CS * 2);

    dim3 blk(256);
    const int nBCL = NB * CS * LSEQ;   // 1,179,648
    const int nMD  = MROWS * DI;       // 2,359,296

    for (int i = 0; i < 4; ++i) {
        prep_kernel<<<(nBCL + 255) / 256, blk, 0, stream>>>(x, prev, xi, uA, i, i > 0 ? 1 : 0);
        cvt_w_kernel<<<(512 * 128 + 255) / 256, blk, 0, stream>>>(Wi + (size_t)i * 512 * 128, wiH, 512, 512, 128);
        cvt_w_kernel<<<(48 * 256 + 255) / 256, blk, 0, stream>>>(Wx + (size_t)i * 40 * 256, wxH, 40, 48, 256);
        cvt_w_kernel<<<(128 * 256 + 255) / 256, blk, 0, stream>>>(Wo + (size_t)i * 128 * 256, woH, 128, 128, 256);

        { // xz = u @ Wi^T : 9216 x 512 x K128 (blocked: 32x64 per wave)
            int waves = (MROWS / 32) * (512 / 64);
            wmma_gemm_blk_kernel<<<(waves + 7) / 8, blk, 0, stream>>>(uA, wiH, xz, MROWS, 512, 128);
        }
        conv_silu_kernel<<<(nMD + 255) / 256, blk, 0, stream>>>(xz, wc + i * DI * 4, bc + i * DI, xs, xsH);
        { // dbc = xs @ Wx^T (N padded to 48) : 9216 x 48 x K256
            int tiles = (MROWS / 16) * (48 / 16);
            wmma_gemm_kernel<<<(tiles + 7) / 8, blk, 0, stream>>>(xsH, wxH, dbc, MROWS, 48, 256);
        }
        delta_kernel<<<(nMD + 255) / 256, blk, 0, stream>>>(dbc, Wdt + i * DI * 8, bdt + i * DI, delta);
        scan_kernel<<<64, blk, 0, stream>>>(delta, xs, dbc, xz, Alog + i * DI * NST, Dp + i * DI, yh);
        { // xm = y @ Wo^T : 9216 x 128 x K256 (blocked: 32x64 per wave)
            int waves = (MROWS / 32) * (128 / 64);
            wmma_gemm_blk_kernel<<<(waves + 7) / 8, blk, 0, stream>>>(yh, woH, xm, MROWS, 128, 256);
        }
        cdgf_reduce_kernel<<<NB * CS, blk, 0, stream>>>(xi, xm, wred);
        cdgf_apply_kernel<<<(nBCL + 255) / 256, blk, 0, stream>>>(xi, xm, wred, out, prev, i);
    }
}
